// MegatronBaselineAttention_67903432950549
// MI455X (gfx1250) — compile-verified, matrix operand order checked by reading, not run on hardware
//
#include <hip/hip_runtime.h>

// ---------------------------------------------------------------------------
// Problem constants (from reference)
// ---------------------------------------------------------------------------
#define SEQ        2048
#define BATCH      2
#define HIDDEN     2048
#define NUM_HEADS  16
#define NUM_KVH    4
#define HEAD_DIM   128
#define Q_PER_KV   4
#define GROUP_SZ   768                 // (Q_PER_KV+2)*HEAD_DIM
#define TOTAL_PROJ 3072                // NUM_KVH*GROUP_SZ
#define ROWS       (SEQ*BATCH)         // 4096
#define SOFTMAX_SCALE 0.08838834764831845f   // 1/sqrt(128)

typedef __attribute__((ext_vector_type(16))) __bf16 v16bf;
typedef __attribute__((ext_vector_type(8)))  float  v8f;
typedef int v4i __attribute__((vector_size(16)));   // matches async-LDS builtin param

#define AS1 __attribute__((address_space(1)))
#define AS3 __attribute__((address_space(3)))

// gfx1250 async memory->LDS path (guarded: fall back to synchronous copies if
// this toolchain does not declare the builtins)
#if defined(__AMDGCN__) && \
    __has_builtin(__builtin_amdgcn_global_load_async_to_lds_b128) && \
    __has_builtin(__builtin_amdgcn_s_wait_asynccnt)
#define HAVE_ASYNC_LDS 1
// src: global int4*, dst: LDS int4*, then imm offset, imm cpol
#define ASYNC_B128(gsrc, ldst) \
  __builtin_amdgcn_global_load_async_to_lds_b128((AS1 v4i*)(gsrc), (AS3 v4i*)(ldst), 0, 0)
#endif

union Frag16 {                 // 16 bf16 values = 8 VGPRs, built from 2x 16B LDS reads
  v16bf v;
  uint4 q[2];
};

__device__ __forceinline__ unsigned short f32_to_bf16(float f) {
  unsigned int u = __float_as_uint(f);
  u += 0x7FFFu + ((u >> 16) & 1u);     // round-to-nearest-even
  return (unsigned short)(u >> 16);
}

__device__ __forceinline__ v8f zero_v8f() {
  v8f z;
#pragma unroll
  for (int i = 0; i < 8; ++i) z[i] = 0.0f;
  return z;
}

__device__ __forceinline__ float red_max16(float v) {
  v = fmaxf(v, __shfl_xor(v, 1, 32));
  v = fmaxf(v, __shfl_xor(v, 2, 32));
  v = fmaxf(v, __shfl_xor(v, 4, 32));
  v = fmaxf(v, __shfl_xor(v, 8, 32));
  return v;
}
__device__ __forceinline__ float red_sum16(float v) {
  v += __shfl_xor(v, 1, 32);
  v += __shfl_xor(v, 2, 32);
  v += __shfl_xor(v, 4, 32);
  v += __shfl_xor(v, 8, 32);
  return v;
}

// ---------------------------------------------------------------------------
// fp32 -> bf16 conversion (vectorized x4)
// ---------------------------------------------------------------------------
__global__ __launch_bounds__(256) void cvt_f32_bf16(const float4* __restrict__ in,
                                                    unsigned short* __restrict__ out,
                                                    int n4) {
  int i = blockIdx.x * 256 + threadIdx.x;
  if (i < n4) {
    float4 f = in[i];
    ushort4 u;
    u.x = f32_to_bf16(f.x); u.y = f32_to_bf16(f.y);
    u.z = f32_to_bf16(f.z); u.w = f32_to_bf16(f.w);
    *(ushort4*)&out[(size_t)i * 4] = u;
  }
}

// ---------------------------------------------------------------------------
// GEMM: C[M][N] (f32) = A[M][K] (bf16) * W[N][K]^T (bf16)
// 128x128 block tile, 256 threads = 8 waves; wave = 32x64 (2x4 WMMA tiles).
// K staged by 64 per barrier (two 32-k WMMA halves) through padded LDS.
// ---------------------------------------------------------------------------
__global__ __launch_bounds__(256) void gemm_bf16_nt(
    const unsigned short* __restrict__ A,
    const unsigned short* __restrict__ W,
    float* __restrict__ C, int M, int N, int K) {
  __shared__ __align__(16) unsigned short As[128 * 72];   // pad 64->72 (16B rows)
  __shared__ __align__(16) unsigned short Bs[128 * 72];

  const int tid   = threadIdx.x;
  const int lane  = tid & 31;
  const int wid   = tid >> 5;
  const int waveM = wid & 3;            // 4 waves along M (4*32 = 128)
  const int waveN = wid >> 2;           // 2 waves along N (2*64 = 128)
  const int hl    = lane >> 4;          // half-wave select
  const int lr    = lane & 15;
  const int bm    = blockIdx.y, bn = blockIdx.x;

  v8f acc[2][4];
#pragma unroll
  for (int tm = 0; tm < 2; ++tm)
#pragma unroll
    for (int tn = 0; tn < 4; ++tn) acc[tm][tn] = zero_v8f();

  for (int k0 = 0; k0 < K; k0 += 64) {
    __syncthreads();
    // stage 128x64 bf16 per tile = 1024 16B-chunks / 256 thr = 4 each
#ifdef HAVE_ASYNC_LDS
#pragma unroll
    for (int i = 0; i < 4; ++i) {
      int c = tid + i * 256;
      int row = c >> 3, col = c & 7;
      ASYNC_B128(&A[(size_t)(bm * 128 + row) * K + k0 + col * 8],
                 &As[row * 72 + col * 8]);
      ASYNC_B128(&W[(size_t)(bn * 128 + row) * K + k0 + col * 8],
                 &Bs[row * 72 + col * 8]);
    }
    __builtin_amdgcn_s_wait_asynccnt(0);
#else
#pragma unroll
    for (int i = 0; i < 4; ++i) {
      int c = tid + i * 256;
      int row = c >> 3, col = c & 7;
      *(uint4*)&As[row * 72 + col * 8] =
          *(const uint4*)&A[(size_t)(bm * 128 + row) * K + k0 + col * 8];
      *(uint4*)&Bs[row * 72 + col * 8] =
          *(const uint4*)&W[(size_t)(bn * 128 + row) * K + k0 + col * 8];
    }
#endif
    if (k0 + 64 < K) {                  // prefetch next stage into caches
      int row = tid >> 1, col = (tid & 1) * 32;
      __builtin_prefetch(&A[(size_t)(bm * 128 + row) * K + k0 + 64 + col], 0, 1);
      __builtin_prefetch(&W[(size_t)(bn * 128 + row) * K + k0 + 64 + col], 0, 1);
    }
    __syncthreads();

#pragma unroll
    for (int kk = 0; kk < 2; ++kk) {    // two 32-k WMMA steps per stage
      Frag16 a[2], b[4];
#pragma unroll
      for (int tm = 0; tm < 2; ++tm) {
        const unsigned short* p = &As[(waveM * 32 + tm * 16 + lr) * 72 + kk * 32 + hl * 8];
        a[tm].q[0] = *(const uint4*)p;              // k = hl*8 + 0..7
        a[tm].q[1] = *(const uint4*)(p + 16);       // k = 16 + hl*8 + 0..7
      }
#pragma unroll
      for (int tn = 0; tn < 4; ++tn) {
        const unsigned short* p = &Bs[(waveN * 64 + tn * 16 + lr) * 72 + kk * 32 + hl * 8];
        b[tn].q[0] = *(const uint4*)p;
        b[tn].q[1] = *(const uint4*)(p + 16);
      }
#pragma unroll
      for (int tm = 0; tm < 2; ++tm)
#pragma unroll
        for (int tn = 0; tn < 4; ++tn)
          acc[tm][tn] = __builtin_amdgcn_wmma_f32_16x16x32_bf16(
              false, a[tm].v, false, b[tn].v, (short)0, acc[tm][tn], false, false);
    }
  }

#pragma unroll
  for (int tm = 0; tm < 2; ++tm)
#pragma unroll
    for (int tn = 0; tn < 4; ++tn)
#pragma unroll
      for (int r = 0; r < 8; ++r) {
        int m = bm * 128 + waveM * 32 + tm * 16 + r + 8 * hl;
        int n = bn * 128 + waveN * 64 + tn * 16 + lr;
        C[(size_t)m * N + n] = acc[tm][tn][r];
      }
}

// ---------------------------------------------------------------------------
// Scatter fused qkv (f32 [ROWS][TOTAL_PROJ]) -> Q/K/V bf16 [b][h][s][d]
// Q pre-scaled by 1/sqrt(HEAD_DIM).
// ---------------------------------------------------------------------------
__global__ __launch_bounds__(256) void scatter_qkv(
    const float* __restrict__ qkv,
    unsigned short* __restrict__ Qb,
    unsigned short* __restrict__ Kb,
    unsigned short* __restrict__ Vb) {
  int i = blockIdx.x * 256 + threadIdx.x;        // ROWS*TOTAL_PROJ threads
  int m = i / TOTAL_PROJ, n = i - m * TOTAL_PROJ;
  int s = m >> 1, b = m & 1;                     // rows are s*BATCH + b
  int g = n / GROUP_SZ, r = n - g * GROUP_SZ;
  float v = qkv[(size_t)i];
  if (r < Q_PER_KV * HEAD_DIM) {
    int h = g * Q_PER_KV + (r >> 7);
    int d = r & 127;
    Qb[(((size_t)(b * NUM_HEADS + h)) * SEQ + s) * HEAD_DIM + d] =
        f32_to_bf16(v * SOFTMAX_SCALE);
  } else if (r < Q_PER_KV * HEAD_DIM + HEAD_DIM) {
    Kb[(((size_t)(b * NUM_KVH + g)) * SEQ + s) * HEAD_DIM + (r - 512)] = f32_to_bf16(v);
  } else {
    Vb[(((size_t)(b * NUM_KVH + g)) * SEQ + s) * HEAD_DIM + (r - 640)] = f32_to_bf16(v);
  }
}

// ---------------------------------------------------------------------------
// Flash attention, causal. Grid: (SEQ/64, NUM_HEADS, BATCH), 128 threads.
// Each wave owns 16 query rows x 128 dims; key blocks of 64.
// ---------------------------------------------------------------------------
__global__ __launch_bounds__(128) void flash_attn(
    const unsigned short* __restrict__ Qb,
    const unsigned short* __restrict__ Kb,
    const unsigned short* __restrict__ Vb,
    unsigned short* __restrict__ attn) {
  __shared__ __align__(16) unsigned short Qs[64 * 136];   // [q][d], pad 128->136
  __shared__ __align__(16) unsigned short Ks[64 * 136];   // [k][d]
  __shared__ __align__(16) unsigned short Vt[128 * 72];   // [d][k], pad 64->72
  __shared__ __align__(16) unsigned short Ps[4][16 * 72]; // per-wave P, [m][k]

  const int tid = threadIdx.x;
  const int lane = tid & 31, wid = tid >> 5;
  const int hl = lane >> 4, lr = lane & 15;
  const int q0 = blockIdx.x * 64;
  const int h  = blockIdx.y;
  const int b  = blockIdx.z;
  const int kvh = h >> 2;

  const unsigned short* Qh = Qb + ((size_t)(b * NUM_HEADS + h)) * SEQ * HEAD_DIM;
  const unsigned short* Kh = Kb + ((size_t)(b * NUM_KVH + kvh)) * SEQ * HEAD_DIM;
  const unsigned short* Vh = Vb + ((size_t)(b * NUM_KVH + kvh)) * SEQ * HEAD_DIM;

  // cooperative Q load: 64 rows x 128 cols = 1024 16B-chunks / 128 thr = 8 each
#ifdef HAVE_ASYNC_LDS
#pragma unroll
  for (int i = 0; i < 8; ++i) {
    int c = tid + i * 128;
    int row = c >> 4, col = c & 15;
    ASYNC_B128(&Qh[(size_t)(q0 + row) * HEAD_DIM + col * 8],
               &Qs[row * 136 + col * 8]);
  }
  __builtin_amdgcn_s_wait_asynccnt(0);
#else
#pragma unroll
  for (int i = 0; i < 8; ++i) {
    int c = tid + i * 128;
    int row = c >> 4, col = c & 15;
    *(uint4*)&Qs[row * 136 + col * 8] =
        *(const uint4*)&Qh[(size_t)(q0 + row) * HEAD_DIM + col * 8];
  }
#endif
  __syncthreads();

  // per-wave Q A-fragments (16 rows x 128 k), 4 k-steps of 32 — kept in VGPRs
  Frag16 qf[4];
#pragma unroll
  for (int ks = 0; ks < 4; ++ks) {
    const unsigned short* p = &Qs[(wid * 16 + lr) * 136 + ks * 32 + hl * 8];
    qf[ks].q[0] = *(const uint4*)p;
    qf[ks].q[1] = *(const uint4*)(p + 16);
  }

  v8f o[8];
#pragma unroll
  for (int t = 0; t < 8; ++t) o[t] = zero_v8f();
  float mrun[8], lrun[8];
#pragma unroll
  for (int r = 0; r < 8; ++r) { mrun[r] = -1e30f; lrun[r] = 0.0f; }

  for (int j0 = 0; j0 <= q0; j0 += 64) {
    __syncthreads();
    // K block [64][128] row-major
#ifdef HAVE_ASYNC_LDS
#pragma unroll
    for (int i = 0; i < 8; ++i) {
      int c = tid + i * 128;
      int row = c >> 4, col = c & 15;
      ASYNC_B128(&Kh[(size_t)(j0 + row) * HEAD_DIM + col * 8],
                 &Ks[row * 136 + col * 8]);
    }
#else
#pragma unroll
    for (int i = 0; i < 8; ++i) {
      int c = tid + i * 128;
      int row = c >> 4, col = c & 15;
      *(uint4*)&Ks[row * 136 + col * 8] =
          *(const uint4*)&Kh[(size_t)(j0 + row) * HEAD_DIM + col * 8];
    }
#endif
    // V block transposed into Vt[d][k]: vectorized global reads, scalar LDS stores
    {
      union { uint4 q; unsigned short u[8]; } vv;
#pragma unroll
      for (int i = 0; i < 8; ++i) {
        int c = tid + i * 128;                 // 1024 16B-chunks
        int k = c >> 4, d8 = c & 15;
        vv.q = *(const uint4*)&Vh[(size_t)(j0 + k) * HEAD_DIM + d8 * 8];
#pragma unroll
        for (int j = 0; j < 8; ++j)
          Vt[(d8 * 8 + j) * 72 + k] = vv.u[j];
      }
    }
#ifdef HAVE_ASYNC_LDS
    __builtin_amdgcn_s_wait_asynccnt(0);
#endif
    __syncthreads();

    // S = Q * K^T  (16 queries x 64 keys per wave) : 4x4 = 16 WMMAs
    v8f st[4];
#pragma unroll
    for (int tn = 0; tn < 4; ++tn) st[tn] = zero_v8f();
#pragma unroll
    for (int tn = 0; tn < 4; ++tn) {
#pragma unroll
      for (int ks = 0; ks < 4; ++ks) {
        Frag16 kb;
        const unsigned short* p = &Ks[(tn * 16 + lr) * 136 + ks * 32 + hl * 8];
        kb.q[0] = *(const uint4*)p;
        kb.q[1] = *(const uint4*)(p + 16);
        st[tn] = __builtin_amdgcn_wmma_f32_16x16x32_bf16(
            false, qf[ks].v, false, kb.v, (short)0, st[tn], false, false);
      }
    }

    // causal mask on the diagonal block only
    if (j0 == q0) {
#pragma unroll
      for (int tn = 0; tn < 4; ++tn)
#pragma unroll
        for (int r = 0; r < 8; ++r) {
          int key = j0 + tn * 16 + lr;
          int qry = q0 + wid * 16 + r + 8 * hl;
          if (key > qry) st[tn][r] = -1e30f;
        }
    }

    // online softmax: per-lane rows are r + 8*hl (cols spread over 16-lane half)
    float alpha[8], mnew[8];
#pragma unroll
    for (int r = 0; r < 8; ++r) {
      float mc = st[0][r];
      mc = fmaxf(mc, st[1][r]); mc = fmaxf(mc, st[2][r]); mc = fmaxf(mc, st[3][r]);
      mc = red_max16(mc);
      mnew[r] = fmaxf(mrun[r], mc);
      alpha[r] = __expf(mrun[r] - mnew[r]);
      mrun[r] = mnew[r];
    }
    float rs[8];
#pragma unroll
    for (int r = 0; r < 8; ++r) rs[r] = 0.0f;
#pragma unroll
    for (int tn = 0; tn < 4; ++tn)
#pragma unroll
      for (int r = 0; r < 8; ++r) {
        float p = __expf(st[tn][r] - mnew[r]);
        st[tn][r] = p;
        rs[r] += p;
      }
#pragma unroll
    for (int r = 0; r < 8; ++r) {
      lrun[r] = lrun[r] * alpha[r] + red_sum16(rs[r]);
#pragma unroll
      for (int t = 0; t < 8; ++t) o[t][r] *= alpha[r];
    }

    // P: D-layout -> A-layout via LDS
#pragma unroll
    for (int tn = 0; tn < 4; ++tn)
#pragma unroll
      for (int r = 0; r < 8; ++r)
        Ps[wid][(r + 8 * hl) * 72 + tn * 16 + lr] = f32_to_bf16(st[tn][r]);
    __syncthreads();

    // O += P * V : 2 k-steps x 8 n-tiles = 16 WMMAs
#pragma unroll
    for (int ks2 = 0; ks2 < 2; ++ks2) {
      Frag16 pf;
      const unsigned short* pp = &Ps[wid][lr * 72 + ks2 * 32 + hl * 8];
      pf.q[0] = *(const uint4*)pp;
      pf.q[1] = *(const uint4*)(pp + 16);
#pragma unroll
      for (int tn = 0; tn < 8; ++tn) {
        Frag16 vf;
        const unsigned short* vp = &Vt[(tn * 16 + lr) * 72 + ks2 * 32 + hl * 8];
        vf.q[0] = *(const uint4*)vp;
        vf.q[1] = *(const uint4*)(vp + 16);
        o[tn] = __builtin_amdgcn_wmma_f32_16x16x32_bf16(
            false, pf.v, false, vf.v, (short)0, o[tn], false, false);
      }
    }
  }

  // epilogue: normalize and store attn [s*BATCH+b][h*128+d] as bf16
#pragma unroll
  for (int t = 0; t < 8; ++t)
#pragma unroll
    for (int r = 0; r < 8; ++r) {
      int s = q0 + wid * 16 + r + 8 * hl;
      int d = t * 16 + lr;
      float val = o[t][r] / lrun[r];
      attn[((size_t)(s * BATCH + b)) * HIDDEN + h * HEAD_DIM + d] = f32_to_bf16(val);
    }
}

// ---------------------------------------------------------------------------
// Host-side orchestration
// ---------------------------------------------------------------------------
extern "C" void kernel_launch(void* const* d_in, const int* in_sizes, int n_in,
                              void* d_out, int out_size, void* d_ws, size_t ws_size,
                              hipStream_t stream) {
  (void)in_sizes; (void)n_in; (void)out_size; (void)ws_size;
  const float* x     = (const float*)d_in[0];   // [SEQ][BATCH][HIDDEN]
  const float* wqkv  = (const float*)d_in[1];   // [TOTAL_PROJ][HIDDEN]
  const float* wproj = (const float*)d_in[2];   // [HIDDEN][HIDDEN]
  float* out = (float*)d_out;

  char* ws = (char*)d_ws;
  const size_t XB    = 0;                                   // 16,777,216 B
  const size_t WQKV  = XB    + (size_t)ROWS * HIDDEN * 2;   // 12,582,912 B
  const size_t WPROJ = WQKV  + (size_t)TOTAL_PROJ * HIDDEN * 2;
  const size_t QKVF  = WPROJ + (size_t)HIDDEN * HIDDEN * 2; // 50,331,648 B
  const size_t QB    = QKVF  + (size_t)ROWS * TOTAL_PROJ * 4;
  const size_t KB    = QB    + (size_t)BATCH * NUM_HEADS * SEQ * HEAD_DIM * 2;
  const size_t VB    = KB    + (size_t)BATCH * NUM_KVH * SEQ * HEAD_DIM * 2;
  const size_t ATTNB = XB;   // alias: x_bf16 is dead after QKV GEMM

  unsigned short* xb    = (unsigned short*)(ws + XB);
  unsigned short* wqkvb = (unsigned short*)(ws + WQKV);
  unsigned short* wprjb = (unsigned short*)(ws + WPROJ);
  float*          qkvf  = (float*)(ws + QKVF);
  unsigned short* Qb    = (unsigned short*)(ws + QB);
  unsigned short* Kb    = (unsigned short*)(ws + KB);
  unsigned short* Vb    = (unsigned short*)(ws + VB);
  unsigned short* attnb = (unsigned short*)(ws + ATTNB);

  // 1. fp32 -> bf16 conversions
  {
    int n4x = ROWS * HIDDEN / 4;
    cvt_f32_bf16<<<(n4x + 255) / 256, 256, 0, stream>>>((const float4*)x, xb, n4x);
    int n4q = TOTAL_PROJ * HIDDEN / 4;
    cvt_f32_bf16<<<(n4q + 255) / 256, 256, 0, stream>>>((const float4*)wqkv, wqkvb, n4q);
    int n4p = HIDDEN * HIDDEN / 4;
    cvt_f32_bf16<<<(n4p + 255) / 256, 256, 0, stream>>>((const float4*)wproj, wprjb, n4p);
  }

  // 2. fused QKV projection: [4096 x 3072] = x_bf16 * wqkv^T
  gemm_bf16_nt<<<dim3(TOTAL_PROJ / 128, ROWS / 128), 256, 0, stream>>>(
      xb, wqkvb, qkvf, ROWS, TOTAL_PROJ, HIDDEN);

  // 3. scatter to Q (pre-scaled) / K / V, bf16 [b][h][s][d]
  scatter_qkv<<<(ROWS * TOTAL_PROJ) / 256, 256, 0, stream>>>(qkvf, Qb, Kb, Vb);

  // 4. causal flash attention -> attn bf16 [s*b][h*d]
  flash_attn<<<dim3(SEQ / 64, NUM_HEADS, BATCH), 128, 0, stream>>>(Qb, Kb, Vb, attnb);

  // 5. output projection: out f32 [4096 x 2048] = attn * wproj^T
  gemm_bf16_nt<<<dim3(HIDDEN / 128, ROWS / 128), 256, 0, stream>>>(
      attnb, wprjb, out, ROWS, HIDDEN, HIDDEN);
}
